// SS2DBlock_18837726560379
// MI455X (gfx1250) — matrix operand order repaired and checked by reading, not compiled, not run on hardware
//
#include <hip/hip_runtime.h>
#include <hip/hip_bf16.h>

typedef __attribute__((ext_vector_type(16))) __bf16 v16bf;
typedef __attribute__((ext_vector_type(8)))  __bf16 v8bf;
typedef __attribute__((ext_vector_type(8)))  float  v8f;

#define NDIR    4
#define DMODEL  512
#define DINNER  1024
#define NROWS   32768
#define ROWS    64          // rows per workgroup
#define XS_STRIDE 520       // bf16 elems (512 + 8 pad) -> 4-dword bank skew
#define ZS_STRIDE 1032      // bf16 elems (1024 + 8 pad)
#define MS_STRIDE 516       // f32 elems  (512 + 4 pad)
#define XS_BYTES  (ROWS * XS_STRIDE * 2)   // 66560
#define ZS_BYTES  (ROWS * ZS_STRIDE * 2)   // 132096
#define RED_BYTES (ROWS * 8 * 4)           // 2048
#define STATS_BYTES (ROWS * 2 * 4)         // 512
#define SMEM_BYTES (XS_BYTES + ZS_BYTES + RED_BYTES + STATS_BYTES)  // 201216

static __device__ __forceinline__ __bf16 f2bf(float f) {
    unsigned u = __builtin_bit_cast(unsigned, f);
    u += 0x7FFFu + ((u >> 16) & 1u);              // round-to-nearest-even
    unsigned short h = (unsigned short)(u >> 16);
    return __builtin_bit_cast(__bf16, h);
}
static __device__ __forceinline__ float bf2f(__bf16 b) {
    unsigned short h = __builtin_bit_cast(unsigned short, b);
    unsigned u = ((unsigned)h) << 16;
    return __builtin_bit_cast(float, u);
}
static __device__ __forceinline__ float sigf(float x) {
    return 1.0f / (1.0f + __expf(-x));
}

// ---------------------------------------------------------------------------
// Prep kernel 1: w_in fp32 [4][512][2048] -> bf16, swizzled into WMMA
// B-fragment layout: per (dir, tileK(32), tileN(16)) tile of 512 elems,
// lane l owns 16 contiguous bf16: K = tk*32 + (l>>4)*16 + j, N = tn*16 + (l&15)
// ---------------------------------------------------------------------------
__global__ __launch_bounds__(256) void k_swz_win(const float* __restrict__ w,
                                                 __bf16* __restrict__ o) {
    unsigned t  = blockIdx.x * 256u + threadIdx.x;       // < 4 * 2^20
    unsigned j  = t & 15u;
    unsigned l  = (t >> 4) & 31u;
    unsigned tn = (t >> 9) & 127u;
    unsigned tk = (t >> 16) & 15u;
    unsigned d  = t >> 20;
    unsigned k  = tk * 32u + (l >> 4) * 16u + j;
    unsigned n  = tn * 16u + (l & 15u);
    o[t] = f2bf(w[(size_t)d * (512u * 2048u) + (size_t)k * 2048u + n]);
}

// ---------------------------------------------------------------------------
// Prep kernel 2: w_comb[d] = w_out[d] (1024x512) @ w_merge[d*512: , :] (512x512)
// computed in f32, written bf16 swizzled (tiles: [32 tk][32 tn][512])
// ---------------------------------------------------------------------------
__global__ __launch_bounds__(256) void k_comb(const float* __restrict__ w_out,
                                              const float* __restrict__ w_merge,
                                              __bf16* __restrict__ o) {
    unsigned t  = blockIdx.x * 256u + threadIdx.x;       // < 2^21
    unsigned j  = t & 15u;
    unsigned l  = (t >> 4) & 31u;
    unsigned tn = (t >> 9) & 31u;
    unsigned tk = (t >> 14) & 31u;
    unsigned d  = t >> 19;
    unsigned k  = tk * 32u + (l >> 4) * 16u + j;
    unsigned n  = tn * 16u + (l & 15u);
    const float* wo = w_out   + (size_t)d * (1024u * 512u) + (size_t)k * 512u;
    const float* wm = w_merge + (size_t)d * (512u * 512u) + n;
    float acc = 0.0f;
    for (int e = 0; e < 512; ++e) acc += wo[e] * wm[(size_t)e * 512u];
    o[t] = f2bf(acc);
}

// ---------------------------------------------------------------------------
// Prep kernel 3: b_eff[n] = b_merge[n] + sum_f b_out_flat[f] * w_merge[f][n]
// ---------------------------------------------------------------------------
__global__ __launch_bounds__(256) void k_beff(const float* __restrict__ b_out,
                                              const float* __restrict__ w_merge,
                                              const float* __restrict__ b_merge,
                                              float* __restrict__ b_eff) {
    int n = blockIdx.x * 256 + threadIdx.x;
    if (n >= 512) return;
    float acc = b_merge[n];
    for (int f = 0; f < 2048; ++f) acc += b_out[f] * w_merge[(size_t)f * 512 + n];
    b_eff[n] = acc;
}

// ---------------------------------------------------------------------------
// Fused SS2D kernel. 256 threads (8 waves), 64 rows per workgroup.
// ---------------------------------------------------------------------------
__global__ __launch_bounds__(256, 1) void ss2d_fused(
    const float* __restrict__ x,        // [32768][512]
    const float* __restrict__ b_in,     // [4][2048]
    const float* __restrict__ ln_g,     // [4][1024]
    const float* __restrict__ ln_b,     // [4][1024]
    const float* __restrict__ norm_g,   // [512]
    const float* __restrict__ norm_b,   // [512]
    const __bf16* __restrict__ win_swz,   // [4][16][128][512]
    const __bf16* __restrict__ wcomb_swz, // [4][32][32][512]
    const float* __restrict__ b_eff,    // [512]
    float* __restrict__ out)            // [32768][512]
{
    extern __shared__ __align__(16) char smem[];
    __bf16* xs    = (__bf16*)smem;                                  // [64][520]
    __bf16* zs    = (__bf16*)(smem + XS_BYTES);                     // [64][1032]
    float*  red   = (float*)(smem + XS_BYTES + ZS_BYTES);           // [64][8]
    float*  stats = (float*)(smem + XS_BYTES + ZS_BYTES + RED_BYTES); // [64][2]
    float*  ms    = (float*)zs;   // aliases zs: zs dead before ms is written

    const int tid  = threadIdx.x;
    const int lane = tid & 31;
    const int wv   = tid >> 5;
    const int lm   = lane & 15;       // N/M index within fragment
    const int lg   = lane >> 4;       // lane-group (K / row-half select)
    const int row0 = blockIdx.x * ROWS;

    // ---- stage x tile: fp32 global -> bf16 LDS (coalesced float4 reads) ----
    for (int it = 0; it < 32; ++it) {
        int idx = it * 1024 + tid * 4;            // < 32768
        int r = idx >> 9, c = idx & 511;
        const float4 v = *(const float4*)(x + (size_t)(row0 + r) * 512 + c);
        __bf16* p = xs + r * XS_STRIDE + c;
        p[0] = f2bf(v.x); p[1] = f2bf(v.y); p[2] = f2bf(v.z); p[3] = f2bf(v.w);
    }
    __syncthreads();

    // persistent merged-output accumulators: wave owns cols [wv*64, wv*64+64)
    v8f accm[4][4];
#pragma unroll
    for (int mt = 0; mt < 4; ++mt)
#pragma unroll
        for (int nt = 0; nt < 4; ++nt) accm[mt][nt] = (v8f){};

    for (int d = 0; d < NDIR; ++d) {
        const __bf16* Wz = win_swz   + (size_t)d * (512u * 2048u);
        const __bf16* Wc = wcomb_swz + (size_t)d * (1024u * 512u);
        const float*  bi = b_in + d * 2048;

        // ================= GEMM1 (+bias, silu*sigmoid) =================
        // wave owns z cols [wv*128, wv*128+128); gate cols at +1024
        for (int nc = 0; nc < 8; ++nc) {
            const int tnz = wv * 8 + nc;        // z-half tile col   [0,64)
            const int tng = 64 + wv * 8 + nc;   // gate-half tile col[64,128)
            v8f az[4], ag[4];
#pragma unroll
            for (int mt = 0; mt < 4; ++mt) { az[mt] = (v8f){}; ag[mt] = (v8f){}; }

#pragma unroll 1
            for (int kt = 0; kt < 16; ++kt) {
                if (kt + 2 < 16) {
                    __builtin_prefetch(Wz + ((size_t)((kt + 2) * 128 + tnz)) * 512 + lane * 16, 0, 3);
                    __builtin_prefetch(Wz + ((size_t)((kt + 2) * 128 + tng)) * 512 + lane * 16, 0, 3);
                }
                v16bf A[4];
                const int koff = kt * 32 + lg * 16;
#pragma unroll
                for (int mt = 0; mt < 4; ++mt) {
                    const __bf16* p = xs + (mt * 16 + lm) * XS_STRIDE + koff;
                    v8bf lo = *(const v8bf*)p;
                    v8bf hi = *(const v8bf*)(p + 8);
                    A[mt] = __builtin_shufflevector(lo, hi,
                        0,1,2,3,4,5,6,7,8,9,10,11,12,13,14,15);
                }
                v16bf Bz = *(const v16bf*)(Wz + ((size_t)(kt * 128 + tnz)) * 512 + lane * 16);
                v16bf Bg = *(const v16bf*)(Wz + ((size_t)(kt * 128 + tng)) * 512 + lane * 16);
#pragma unroll
                for (int mt = 0; mt < 4; ++mt) {
                    az[mt] = __builtin_amdgcn_wmma_f32_16x16x32_bf16(
                        false, A[mt], false, Bz, (short)0, az[mt], false, false);
                    ag[mt] = __builtin_amdgcn_wmma_f32_16x16x32_bf16(
                        false, A[mt], false, Bg, (short)0, ag[mt], false, false);
                }
            }
            // activation + store bf16 z to LDS (C layout: vgpr i -> M = lg*8+i)
            const int cz = tnz * 16 + lm;            // z column in [0,1024)
            const float bz = bi[cz];
            const float bg = bi[1024 + cz];
#pragma unroll
            for (int mt = 0; mt < 4; ++mt) {
#pragma unroll
                for (int i = 0; i < 8; ++i) {
                    float zv = az[mt][i] + bz;
                    float gv = ag[mt][i] + bg;
                    float s  = zv * sigf(zv) * sigf(gv);   // silu(z)*sigmoid(g)
                    int rr = mt * 16 + lg * 8 + i;
                    zs[rr * ZS_STRIDE + cz] = f2bf(s);
                }
            }
        }
        __syncthreads();

        // ================= LayerNorm over 1024 (4 threads / row) ==========
        {
            const int r = tid >> 2, q = tid & 3;
            const __bf16* zr = zs + r * ZS_STRIDE + q * 256;
            float s = 0.0f, s2 = 0.0f;
#pragma unroll 4
            for (int c = 0; c < 256; c += 8) {
                v8bf v = *(const v8bf*)(zr + c);
#pragma unroll
                for (int i = 0; i < 8; ++i) { float f = bf2f(v[i]); s += f; s2 += f * f; }
            }
            red[r * 8 + q] = s;
            red[r * 8 + 4 + q] = s2;
        }
        __syncthreads();
        if (tid < ROWS) {
            float su = red[tid*8+0] + red[tid*8+1] + red[tid*8+2] + red[tid*8+3];
            float sq = red[tid*8+4] + red[tid*8+5] + red[tid*8+6] + red[tid*8+7];
            float mu  = su * (1.0f / 1024.0f);
            float var = sq * (1.0f / 1024.0f) - mu * mu;
            stats[tid * 2]     = mu;
            stats[tid * 2 + 1] = rsqrtf(var + 1e-5f);
        }
        __syncthreads();
        {
            const int r = tid >> 2, q = tid & 3;
            const float mu   = stats[r * 2];
            const float rstd = stats[r * 2 + 1];
            __bf16* zr = zs + r * ZS_STRIDE + q * 256;
            const float* lgp = ln_g + d * 1024 + q * 256;
            const float* lbp = ln_b + d * 1024 + q * 256;
#pragma unroll 2
            for (int c = 0; c < 256; c += 8) {
                v8bf v = *(const v8bf*)(zr + c);
                v8bf o;
#pragma unroll
                for (int i = 0; i < 8; ++i)
                    o[i] = f2bf((bf2f(v[i]) - mu) * rstd * lgp[c + i] + lbp[c + i]);
                *(v8bf*)(zr + c) = o;
            }
        }
        __syncthreads();

        // ================= GEMM2: accm += z_norm @ w_comb[d] ==============
#pragma unroll 1
        for (int kt = 0; kt < 32; ++kt) {
            if (kt + 2 < 32) {
                __builtin_prefetch(Wc + ((size_t)((kt + 2) * 32 + wv * 4)) * 512 + lane * 16, 0, 3);
            }
            v16bf A[4];
            const int koff = kt * 32 + lg * 16;
#pragma unroll
            for (int mt = 0; mt < 4; ++mt) {
                const __bf16* p = zs + (mt * 16 + lm) * ZS_STRIDE + koff;
                v8bf lo = *(const v8bf*)p;
                v8bf hi = *(const v8bf*)(p + 8);
                A[mt] = __builtin_shufflevector(lo, hi,
                    0,1,2,3,4,5,6,7,8,9,10,11,12,13,14,15);
            }
            v16bf Bc[4];
#pragma unroll
            for (int nt = 0; nt < 4; ++nt)
                Bc[nt] = *(const v16bf*)(Wc + ((size_t)(kt * 32 + wv * 4 + nt)) * 512 + lane * 16);
#pragma unroll
            for (int mt = 0; mt < 4; ++mt)
#pragma unroll
                for (int nt = 0; nt < 4; ++nt)
                    accm[mt][nt] = __builtin_amdgcn_wmma_f32_16x16x32_bf16(
                        false, A[mt], false, Bc[nt], (short)0, accm[mt][nt], false, false);
        }
        __syncthreads();   // zs dead; next direction may overwrite it
    }

    // ---- spill merged accumulators (+b_eff) to LDS (ms aliases zs) -------
#pragma unroll
    for (int mt = 0; mt < 4; ++mt) {
#pragma unroll
        for (int nt = 0; nt < 4; ++nt) {
            const int cc = wv * 64 + nt * 16 + lm;
            const float be = b_eff[cc];
#pragma unroll
            for (int i = 0; i < 8; ++i) {
                int rr = mt * 16 + lg * 8 + i;
                ms[rr * MS_STRIDE + cc] = accm[mt][nt][i] + be;
            }
        }
    }
    __syncthreads();

    // ---- final LayerNorm over 512 + scale/shift + store ------------------
    {
        const int r = tid >> 2, q = tid & 3;
        const float* mr = ms + r * MS_STRIDE + q * 128;
        float s = 0.0f, s2 = 0.0f;
#pragma unroll 4
        for (int c = 0; c < 128; ++c) { float v = mr[c]; s += v; s2 += v * v; }
        red[r * 8 + q] = s;
        red[r * 8 + 4 + q] = s2;
    }
    __syncthreads();
    if (tid < ROWS) {
        float su = red[tid*8+0] + red[tid*8+1] + red[tid*8+2] + red[tid*8+3];
        float sq = red[tid*8+4] + red[tid*8+5] + red[tid*8+6] + red[tid*8+7];
        float mu  = su * (1.0f / 512.0f);
        float var = sq * (1.0f / 512.0f) - mu * mu;
        stats[tid * 2]     = mu;
        stats[tid * 2 + 1] = rsqrtf(var + 1e-5f);
    }
    __syncthreads();
    {
        const int r = tid >> 2, q = tid & 3;
        const float mu   = stats[r * 2];
        const float rstd = stats[r * 2 + 1];
        const float* mr = ms + r * MS_STRIDE;
        float* op = out + (size_t)(row0 + r) * 512;
#pragma unroll 2
        for (int c = q * 128; c < q * 128 + 128; c += 4) {
            float4 o;
            o.x = (mr[c+0] - mu) * rstd * norm_g[c+0] + norm_b[c+0];
            o.y = (mr[c+1] - mu) * rstd * norm_g[c+1] + norm_b[c+1];
            o.z = (mr[c+2] - mu) * rstd * norm_g[c+2] + norm_b[c+2];
            o.w = (mr[c+3] - mu) * rstd * norm_g[c+3] + norm_b[c+3];
            *(float4*)(op + c) = o;
        }
    }
}

// ---------------------------------------------------------------------------
extern "C" void kernel_launch(void* const* d_in, const int* in_sizes, int n_in,
                              void* d_out, int out_size, void* d_ws, size_t ws_size,
                              hipStream_t stream) {
    const float* x       = (const float*)d_in[0];
    const float* w_in    = (const float*)d_in[1];
    const float* b_in    = (const float*)d_in[2];
    const float* ln_g    = (const float*)d_in[3];
    const float* ln_b    = (const float*)d_in[4];
    const float* w_out   = (const float*)d_in[5];
    const float* b_out   = (const float*)d_in[6];
    const float* w_merge = (const float*)d_in[7];
    const float* b_merge = (const float*)d_in[8];
    const float* norm_g  = (const float*)d_in[9];
    const float* norm_b  = (const float*)d_in[10];
    float* outp = (float*)d_out;

    char* ws = (char*)d_ws;
    __bf16* win_swz   = (__bf16*)ws;                         // 8 MB
    __bf16* wcomb_swz = (__bf16*)(ws + (size_t)(8u << 20));  // 4 MB
    float*  b_eff     = (float*)(ws + (size_t)(12u << 20));  // 2 KB

    // weight preprocessing (bf16 convert + WMMA-fragment swizzle + folding)
    k_swz_win<<<(4u << 20) / 256, 256, 0, stream>>>(w_in, win_swz);
    k_comb<<<(2u << 20) / 256, 256, 0, stream>>>(w_out, w_merge, wcomb_swz);
    k_beff<<<2, 256, 0, stream>>>(b_out, w_merge, b_merge, b_eff);

    (void)hipFuncSetAttribute((const void*)ss2d_fused,
                              hipFuncAttributeMaxDynamicSharedMemorySize,
                              (int)SMEM_BYTES);
    ss2d_fused<<<NROWS / ROWS, 256, SMEM_BYTES, stream>>>(
        x, b_in, ln_g, ln_b, norm_g, norm_b, win_swz, wcomb_swz, b_eff, outp);
}
